// WindowAttention_79465484911019
// MI455X (gfx1250) — matrix operand order repaired and checked by reading, not compiled, not run on hardware
//
#include <hip/hip_runtime.h>
#include <cstddef>

typedef __bf16 bf16;
typedef __attribute__((ext_vector_type(16))) __bf16 v16bf;
typedef __attribute__((ext_vector_type(8)))  __bf16 v8bf;
typedef __attribute__((ext_vector_type(4)))  __bf16 v4bf;
typedef __attribute__((ext_vector_type(8)))  float   v8f;
typedef __attribute__((ext_vector_type(4)))  float   v4f;

#define NHD   16      // heads
#define HD    32      // head dim
#define CDIM  512     // channels
#define NTOK  64      // tokens per 8x8 window
#define LDA   520     // padded row stride (bf16 elems) for 64x512 LDS tiles (mult of 8 -> 16B aligned rows)
#define LDV   72      // padded token stride for V^T (512 x 64)
#define LDSC  72      // scores row stride (16 x 64 per wave)

static __device__ __forceinline__ v16bf cat16(v8bf lo, v8bf hi) {
  return __builtin_shufflevector(lo, hi, 0,1,2,3,4,5,6,7,8,9,10,11,12,13,14,15);
}
static __device__ __forceinline__ v8f wmma_bf16(v16bf a, v16bf b, v8f c) {
  // D = A(16x32 bf16) x B(32x16 bf16) + C(16x16 f32)
  return __builtin_amdgcn_wmma_f32_16x16x32_bf16(false, a, false, b, (short)0, c, false, false);
}

// ---------------------------------------------------------------------------
// Kernel 1: one-time weight conversion f32 -> bf16, pre-packed in WMMA
// B-fragment order. Fragment (ntile, kblock) stores 512 bf16 contiguously:
// dst[lane*16 + e] = W[(kb*32 + e + 16*(lane>>4)) , ntile*16 + (lane&15)]
// fragIdx = ntile*16 + kb  (kb innermost => streaming inner GEMM loop)
// ---------------------------------------------------------------------------
__global__ __launch_bounds__(256) void swin_weight_swizzle(
    const float* __restrict__ qkv_w,   // (512, 1536)
    const float* __restrict__ proj_w,  // (512, 512)
    bf16* __restrict__ qkvsw, bf16* __restrict__ projsw)
{
  const int lane = threadIdx.x & 31, wv = threadIdx.x >> 5;
  const int ll = lane & 15, lh = lane >> 4;
  const int gw = blockIdx.x * 8 + wv;          // 0..2047 waves, one fragment each
  if (gw < 1536) {                             // qkv_w: 96 ntiles x 16 kblocks
    const int nt = gw >> 4, kb = gw & 15;
    bf16* dst = qkvsw + (size_t)gw * 512 + lane * 16;
    const float* src = qkv_w + (size_t)(kb * 32 + 16 * lh) * 1536 + nt * 16 + ll;
#pragma unroll
    for (int e = 0; e < 16; ++e) dst[e] = (bf16)src[(size_t)e * 1536];
  } else {                                     // proj_w: 32 ntiles x 16 kblocks
    const int p = gw - 1536;
    const int nt = p >> 4, kb = p & 15;
    bf16* dst = projsw + (size_t)p * 512 + lane * 16;
    const float* src = proj_w + (size_t)(kb * 32 + 16 * lh) * 512 + nt * 16 + ll;
#pragma unroll
    for (int e = 0; e < 16; ++e) dst[e] = (bf16)src[(size_t)e * 512];
  }
}

// ---------------------------------------------------------------------------
// Kernel 2: fully fused window attention. One block (8 wave32) per window.
// LDS: x-tile (reused as attn-out), Q, K, V^T, per-wave score buffers.
// ---------------------------------------------------------------------------
__global__ __launch_bounds__(256) void swin_attn_fused(
    const float* __restrict__ x,        // (8,128,128,512)
    const float* __restrict__ qkv_b,    // (1536)
    const float* __restrict__ proj_b,   // (512)
    const float* __restrict__ bias_tab, // (225, 16)
    const bf16* __restrict__ qkvsw,     // swizzled (512,1536) bf16
    const bf16* __restrict__ projsw,    // swizzled (512,512)  bf16
    float* __restrict__ out)            // (8,128,128,512)
{
  __shared__ bf16 s_xw[NTOK * LDA];     // window input; reused as attn output
  __shared__ bf16 s_q [NTOK * LDA];     // Q (pre-scaled), row-major [tok][c]
  __shared__ bf16 s_k [NTOK * LDA];     // K, row-major [tok][c]
  __shared__ bf16 s_vT[CDIM * LDV];     // V transposed [c][tok]
  __shared__ bf16 s_sc[8 * 16 * LDSC];  // per-wave softmax scores 16x64

  const int tid  = threadIdx.x;
  const int lane = tid & 31, wv = tid >> 5;
  const int ll   = lane & 15, lh = lane >> 4;

  const int blk = blockIdx.x;                 // 2048 windows
  const int b   = blk >> 8;
  const int gy0 = ((blk >> 4) & 15) * 8;
  const int gx0 = (blk & 15) * 8;

  // ---- Phase 0: window partition, f32 -> bf16 into LDS ----
#pragma unroll
  for (int i = 0; i < 32; ++i) {
    int idx = tid + i * 256;                  // float4 index, 8192 total
    int row = idx >> 7;                       // token 0..63
    int c4  = (idx & 127) << 2;               // channel
    int gy = gy0 + (row >> 3), gx = gx0 + (row & 7);
    v4f v = *(const v4f*)(x + (((size_t)b * 128 + gy) * 128 + gx) * 512 + c4);
    v4bf h; h[0] = (bf16)v[0]; h[1] = (bf16)v[1]; h[2] = (bf16)v[2]; h[3] = (bf16)v[3];
    *(v4bf*)(&s_xw[row * LDA + c4]) = h;
  }
  __syncthreads();

  // ---- Phase 1: QKV GEMM (64x512)@(512x1536), write Q/K/V^T to LDS ----
  const float scale = 0.17677669529663687f;   // 32^-0.5 folded into Q
  for (int t = wv; t < 384; t += 8) {         // 4 mtiles x 96 ntiles
    const int mt = t & 3, nt = t >> 2;
    v8f acc = {};
    const bf16* arow = s_xw + (mt * 16 + ll) * LDA + 16 * lh;
    const bf16* bfr  = qkvsw + (size_t)(nt * 16) * 512 + lane * 16;
#pragma unroll
    for (int kb = 0; kb < 16; ++kb) {
      v8bf a0 = *(const v8bf*)(arow + kb * 32);
      v8bf a1 = *(const v8bf*)(arow + kb * 32 + 8);
      v8bf b0 = *(const v8bf*)(bfr + (size_t)kb * 512);
      v8bf b1 = *(const v8bf*)(bfr + (size_t)kb * 512 + 8);
      acc = wmma_bf16(cat16(a0, a1), cat16(b0, b1), acc);
    }
    const int ncol = nt * 16 + ll;            // 0..1535 (branch uniform per wave)
    const float bia = qkv_b[ncol];
    if (ncol < 512) {
#pragma unroll
      for (int r = 0; r < 8; ++r)
        s_q[(mt * 16 + r + 8 * lh) * LDA + ncol] = (bf16)((acc[r] + bia) * scale);
    } else if (ncol < 1024) {
      const int c = ncol - 512;
#pragma unroll
      for (int r = 0; r < 8; ++r)
        s_k[(mt * 16 + r + 8 * lh) * LDA + c] = (bf16)(acc[r] + bia);
    } else {
      const int c = ncol - 1024;
#pragma unroll
      for (int r = 0; r < 8; ++r)
        s_vT[c * LDV + (mt * 16 + r + 8 * lh)] = (bf16)(acc[r] + bia);
    }
  }
  __syncthreads();

  // ---- Phase 2: attention, one (head, mtile) pair per wave-iteration ----
  bf16* mysc = s_sc + wv * 16 * LDSC;
  for (int p = wv; p < 64; p += 8) {
    const int h = p >> 2, mt = p & 3;
    // scores = (q*scale) . k^T : 4 tiles of 16x16, K = head_dim = 32 (one WMMA)
    const bf16* aq = s_q + (mt * 16 + ll) * LDA + h * HD + 16 * lh;
    v16bf aQ = cat16(*(const v8bf*)aq, *(const v8bf*)(aq + 8));
    v8f sacc[4];
#pragma unroll
    for (int nt = 0; nt < 4; ++nt) {
      const bf16* bk = s_k + (nt * 16 + ll) * LDA + h * HD + 16 * lh;
      v16bf bK = cat16(*(const v8bf*)bk, *(const v8bf*)(bk + 8));
      v8f z = {};
      sacc[nt] = wmma_bf16(aQ, bK, z);
    }
    // + relative position bias, then rowwise softmax over the 64 keys.
    float rmax[8];
#pragma unroll
    for (int r = 0; r < 8; ++r) {
      int qt = mt * 16 + r + 8 * lh;
      int qy = qt >> 3, qx = qt & 7;
      float m = -3.0e38f;
#pragma unroll
      for (int nt = 0; nt < 4; ++nt) {
        int kt = nt * 16 + ll;
        int idx = (qy - (kt >> 3) + 7) * 15 + (qx - (kt & 7) + 7);
        float s = sacc[nt][r] + bias_tab[idx * NHD + h];
        sacc[nt][r] = s;
        m = fmaxf(m, s);
      }
#pragma unroll
      for (int off = 1; off < 16; off <<= 1)   // rows live in one 16-lane half
        m = fmaxf(m, __shfl_xor(m, off, 32));
      rmax[r] = m;
    }
#pragma unroll
    for (int r = 0; r < 8; ++r) {
      float s = 0.f;
#pragma unroll
      for (int nt = 0; nt < 4; ++nt) {
        float e = __expf(sacc[nt][r] - rmax[r]);
        sacc[nt][r] = e;
        s += e;
      }
#pragma unroll
      for (int off = 1; off < 16; off <<= 1)
        s += __shfl_xor(s, off, 32);
      const float inv = 1.0f / s;
      const int lr = r + 8 * lh;               // local row 0..15
#pragma unroll
      for (int nt = 0; nt < 4; ++nt)
        mysc[lr * LDSC + nt * 16 + ll] = (bf16)(sacc[nt][r] * inv);
    }
    // out = attn . v : 16x32 result, K = 64 keys (two WMMA steps)
    const bf16* asc = mysc + ll * LDSC + 16 * lh;
#pragma unroll
    for (int nt2 = 0; nt2 < 2; ++nt2) {
      v8f oacc = {};
#pragma unroll
      for (int kb = 0; kb < 2; ++kb) {
        v16bf aP = cat16(*(const v8bf*)(asc + kb * 32),
                         *(const v8bf*)(asc + kb * 32 + 8));
        const bf16* bv = s_vT + (h * HD + nt2 * 16 + ll) * LDV + kb * 32 + 16 * lh;
        v16bf bV = cat16(*(const v8bf*)bv, *(const v8bf*)(bv + 8));
        oacc = wmma_bf16(aP, bV, oacc);
      }
#pragma unroll
      for (int r = 0; r < 8; ++r)
        s_xw[(mt * 16 + r + 8 * lh) * LDA + h * HD + nt2 * 16 + ll] = (bf16)oacc[r];
    }
  }
  __syncthreads();

  // ---- Phase 3: projection (64x512)@(512x512) + bias + window reverse ----
  for (int t = wv; t < 128; t += 8) {          // 4 mtiles x 32 ntiles
    const int mt = t & 3, nt = t >> 2;
    v8f acc = {};
    const bf16* arow = s_xw + (mt * 16 + ll) * LDA + 16 * lh;
    const bf16* bfr  = projsw + (size_t)(nt * 16) * 512 + lane * 16;
#pragma unroll
    for (int kb = 0; kb < 16; ++kb) {
      v8bf a0 = *(const v8bf*)(arow + kb * 32);
      v8bf a1 = *(const v8bf*)(arow + kb * 32 + 8);
      v8bf b0 = *(const v8bf*)(bfr + (size_t)kb * 512);
      v8bf b1 = *(const v8bf*)(bfr + (size_t)kb * 512 + 8);
      acc = wmma_bf16(cat16(a0, a1), cat16(b0, b1), acc);
    }
    const int c = nt * 16 + ll;
    const float pb = proj_b[c];
#pragma unroll
    for (int r = 0; r < 8; ++r) {
      int tok = mt * 16 + r + 8 * lh;
      int gy = gy0 + (tok >> 3), gx = gx0 + (tok & 7);
      out[(((size_t)b * 128 + gy) * 128 + gx) * 512 + c] = acc[r] + pb;
    }
  }
}

// ---------------------------------------------------------------------------
extern "C" void kernel_launch(void* const* d_in, const int* in_sizes, int n_in,
                              void* d_out, int out_size, void* d_ws, size_t ws_size,
                              hipStream_t stream) {
  (void)in_sizes; (void)n_in; (void)out_size; (void)ws_size;
  const float* x      = (const float*)d_in[0];
  const float* qkv_w  = (const float*)d_in[1];
  const float* qkv_b  = (const float*)d_in[2];
  const float* proj_w = (const float*)d_in[3];
  const float* proj_b = (const float*)d_in[4];
  const float* btab   = (const float*)d_in[5];

  bf16* qkvsw  = (bf16*)d_ws;                       // 1536*512 bf16 = 1.5 MB
  bf16* projsw = qkvsw + (size_t)1536 * 512;        //  512*512 bf16 = 0.5 MB

  swin_weight_swizzle<<<256, 256, 0, stream>>>(qkv_w, proj_w, qkvsw, projsw);
  swin_attn_fused<<<2048, 256, 0, stream>>>(x, qkv_b, proj_b, btab,
                                            qkvsw, projsw, (float*)d_out);
}
